// MoELayer_57861799412422
// MI455X (gfx1250) — compile-verified, matrix operand order checked by reading, not run on hardware
//
#include <hip/hip_runtime.h>
#include <math.h>

// ---------------- problem constants (from reference setup) ----------------
#define BB 4
#define NN 2048
#define DD 1024
#define EE 8
#define HH 4096
#define CC 320            // int(N * 1.25 // E)

// ---------------- WMMA vector types ----------------
typedef __attribute__((ext_vector_type(16))) __bf16          v16bf;
typedef __attribute__((ext_vector_type(8)))  float           v8f;
typedef __attribute__((ext_vector_type(8)))  unsigned short  v8us;
typedef __attribute__((ext_vector_type(16))) unsigned short  v16us;
typedef __attribute__((ext_vector_type(4)))  unsigned short  v4us;

// fp32 -> bf16 round-to-nearest-even (raw bits)
static __device__ inline unsigned short f2bf(float f) {
    unsigned u = __builtin_bit_cast(unsigned, f);
    unsigned r = (u + 0x7FFFu + ((u >> 16) & 1u)) >> 16;
    return (unsigned short)r;
}

// Load one 16x32 bf16 WMMA fragment slice for this lane.
// p points at (row_base + kbase) where kbase = (lane<16?0:8).
// Elements 0..7 <- K[kbase..kbase+7], 8..15 <- K[kbase+16..kbase+23]: two 16B loads.
static __device__ inline v16bf load_frag(const unsigned short* p) {
    v8us lo = *(const v8us*)(p);
    v8us hi = *(const v8us*)(p + 16);
    v16us t = __builtin_shufflevector(lo, hi, 0,1,2,3,4,5,6,7,8,9,10,11,12,13,14,15);
    return __builtin_bit_cast(v16bf, t);
}

static __device__ inline float gelu_tanh(float x) {
    const float k0 = 0.7978845608028654f, k1 = 0.044715f;
    float t = tanhf(k0 * (x + k1 * x * x * x));
    return 0.5f * x * (1.0f + t);
}

// ---------------- init kernels ----------------
__global__ void zero_f32(float* p, int n) {
    int i = blockIdx.x * blockDim.x + threadIdx.x;
    if (i < n) p[i] = 0.0f;
}

__global__ void init_ws_k(int* tok_slot, float* proxy, int* count, float* zz) {
    int i = blockIdx.x * blockDim.x + threadIdx.x;
    if (i < BB * EE * CC) tok_slot[i] = -1;
    if (i < BB * EE) { proxy[i] = 0.0f; count[i] = 0; }
    if (i == 0) *zz = 0.0f;
}

// ---------------- weight transpose + fp32->bf16 convert ----------------
// src: [E][R][Cc] f32  ->  dst: [E][Cc][R] bf16  (per-expert 32x32 LDS-tiled)
__global__ __launch_bounds__(256) void transpose_cvt(const float* __restrict__ src,
                                                     unsigned short* __restrict__ dst,
                                                     int R, int Cc) {
    __shared__ float tile[32][33];
    int tPerRow   = Cc >> 5;
    int tilesPerE = (R >> 5) * tPerRow;
    int e  = blockIdx.x / tilesPerE;
    int t  = blockIdx.x % tilesPerE;
    int r0 = (t / tPerRow) << 5;
    int c0 = (t % tPerRow) << 5;
    int tx = threadIdx.x & 31, ty = threadIdx.x >> 5;   // 32x8 threads
    const float* s = src + (size_t)e * R * Cc;
#pragma unroll
    for (int i = 0; i < 32; i += 8)
        tile[ty + i][tx] = s[(size_t)(r0 + ty + i) * Cc + (c0 + tx)];
    __syncthreads();
    unsigned short* d = dst + (size_t)e * R * Cc;
#pragma unroll
    for (int i = 0; i < 32; i += 8)
        d[(size_t)(c0 + ty + i) * R + (r0 + tx)] = f2bf(tile[tx][ty + i]);
}

// ---------------- router: one wave per token ----------------
__global__ __launch_bounds__(256) void router_k(const float* __restrict__ x,
                                                const float* __restrict__ wr,
                                                int* __restrict__ idx_ws,
                                                float* __restrict__ gate_ws,
                                                float* __restrict__ proxy,
                                                float* __restrict__ zz) {
    int wave = (blockIdx.x * blockDim.x + threadIdx.x) >> 5;
    int lane = threadIdx.x & 31;
    int b = wave / NN;
    const float* xr = x + (size_t)wave * DD;
    float acc[EE];
#pragma unroll
    for (int e = 0; e < EE; ++e) acc[e] = 0.0f;
    for (int d = lane; d < DD; d += 32) {
        float xv = xr[d];
        const float* w = wr + d * EE;
#pragma unroll
        for (int e = 0; e < EE; ++e) acc[e] += xv * w[e];
    }
#pragma unroll
    for (int e = 0; e < EE; ++e)
#pragma unroll
        for (int off = 16; off > 0; off >>= 1)
            acc[e] += __shfl_xor(acc[e], off, 32);
    // all lanes now hold full logits
    float m = acc[0]; int im = 0;
#pragma unroll
    for (int e = 1; e < EE; ++e) if (acc[e] > m) { m = acc[e]; im = e; }
    float s = 0.0f;
#pragma unroll
    for (int e = 0; e < EE; ++e) s += expf(acc[e] - m);
    float z = m + logf(s);
    float gate = 1.0f / s;                 // exp(max - max)/sum
    if (lane < EE) atomicAdd(&proxy[b * EE + lane], expf(acc[lane] - m) / s);
    if (lane == 0) {
        idx_ws[wave]  = im;
        gate_ws[wave] = gate;
        atomicAdd(zz, z * z);
    }
}

// ---------------- capacity scan: one wave per (b,e) ----------------
__global__ __launch_bounds__(32) void scan_k(const int* __restrict__ idx_ws,
                                             const float* __restrict__ gate_ws,
                                             int* __restrict__ tok_slot,
                                             float* __restrict__ slot_gate,
                                             int* __restrict__ count) {
    int b = blockIdx.x >> 3, e = blockIdx.x & 7;
    int lane = threadIdx.x;
    int base = 0;
    for (int n0 = 0; n0 < NN; n0 += 32) {
        int n = n0 + lane;
        bool match = (idx_ws[b * NN + n] == e);
        unsigned mask = (unsigned)__ballot(match);
        int prefix = __popc(mask & ((1u << lane) - 1u));
        if (match) {
            int p = base + prefix;
            if (p < CC) {
                int sidx = (b * EE + e) * CC + p;
                tok_slot[sidx]  = n;
                slot_gate[sidx] = gate_ws[b * NN + n];
            }
        }
        base += __popc(mask);
    }
    if (lane == 0) count[b * EE + e] = base;
}

// ---------------- gather tokens into bf16 expert-input buffer ----------------
__global__ __launch_bounds__(256) void gather_k(const float* __restrict__ x,
                                                const int* __restrict__ tok_slot,
                                                unsigned short* __restrict__ xe) {
    int row = blockIdx.x;                 // 0 .. B*E*C-1
    int tok = tok_slot[row];
    unsigned short* dst = xe + (size_t)row * DD;
    int d0 = threadIdx.x * 4;
    if (tok < 0) {
        v4us z = {0, 0, 0, 0};
        *(v4us*)(dst + d0) = z;
    } else {
        int b = row / (EE * CC);
        const float* src = x + ((size_t)b * NN + tok) * DD;
        float4 v = *(const float4*)(src + d0);
        v4us o = { f2bf(v.x), f2bf(v.y), f2bf(v.z), f2bf(v.w) };
        *(v4us*)(dst + d0) = o;
    }
}

// ---------------- GEMM1: xe[be] (C x D) @ w1t[e] (H x D)^T -> gelu -> h bf16 ----------------
// per wave: 2 M-tiles x 4 N-tiles (32x64 output), K stepped by 32
__global__ __launch_bounds__(256) void gemm1_k(const unsigned short* __restrict__ xe,
                                               const unsigned short* __restrict__ w1t,
                                               const float* __restrict__ b1,
                                               unsigned short* __restrict__ hbuf) {
    const int MP = CC / 32;          // 10
    const int NS = HH / 64;          // 64
    int wave = (blockIdx.x * blockDim.x + threadIdx.x) >> 5;
    int lane = threadIdx.x & 31;
    int be  = wave / (MP * NS);
    int rem = wave % (MP * NS);
    int mp = rem / NS, ns = rem % NS;
    int e  = be & (EE - 1);
    int m0 = mp * 32, h0 = ns * 64;
    int lm = lane & 15;
    int kbase = (lane < 16) ? 0 : 8;

    const unsigned short* A  = xe  + (size_t)be * CC * DD;
    const unsigned short* Bw = w1t + (size_t)e * HH * DD;
    const unsigned short* arow0 = A + (size_t)(m0 + lm) * DD + kbase;
    const unsigned short* arow1 = arow0 + (size_t)16 * DD;
    const unsigned short* brow[4];
#pragma unroll
    for (int ni = 0; ni < 4; ++ni)
        brow[ni] = Bw + (size_t)(h0 + ni * 16 + lm) * DD + kbase;

    v8f acc[2][4] = {};
    for (int kk = 0; kk < DD; kk += 32) {
        v16bf a0 = load_frag(arow0 + kk);
        v16bf a1 = load_frag(arow1 + kk);
        v16bf bf[4];
#pragma unroll
        for (int ni = 0; ni < 4; ++ni) bf[ni] = load_frag(brow[ni] + kk);
#pragma unroll
        for (int ni = 0; ni < 4; ++ni) {
            acc[0][ni] = __builtin_amdgcn_wmma_f32_16x16x32_bf16(
                false, a0, false, bf[ni], (short)0, acc[0][ni], false, false);
            acc[1][ni] = __builtin_amdgcn_wmma_f32_16x16x32_bf16(
                false, a1, false, bf[ni], (short)0, acc[1][ni], false, false);
        }
    }
    int rowoff = (lane < 16) ? 0 : 8;
#pragma unroll
    for (int mi = 0; mi < 2; ++mi)
#pragma unroll
        for (int ni = 0; ni < 4; ++ni) {
            int hcol = h0 + ni * 16 + lm;
            float bias = b1[e * HH + hcol];
            int c0 = m0 + mi * 16 + rowoff;
#pragma unroll
            for (int r = 0; r < 8; ++r) {
                float v = gelu_tanh(acc[mi][ni][r] + bias);
                hbuf[((size_t)be * CC + (c0 + r)) * HH + hcol] = f2bf(v);
            }
        }
}

// ---------------- GEMM2: h[be] (C x H) @ w2t[e] (D x H)^T -> scale by gate, scatter ----------------
__global__ __launch_bounds__(256) void gemm2_k(const unsigned short* __restrict__ hbuf,
                                               const unsigned short* __restrict__ w2t,
                                               const float* __restrict__ b2,
                                               const int* __restrict__ tok_slot,
                                               const float* __restrict__ slot_gate,
                                               float* __restrict__ out) {
    const int MP = CC / 32;          // 10
    const int NS = DD / 64;          // 16
    int wave = (blockIdx.x * blockDim.x + threadIdx.x) >> 5;
    int lane = threadIdx.x & 31;
    int be  = wave / (MP * NS);
    int rem = wave % (MP * NS);
    int mp = rem / NS, ns = rem % NS;
    int e  = be & (EE - 1);
    int b  = be >> 3;
    int m0 = mp * 32, d0 = ns * 64;
    int lm = lane & 15;
    int kbase = (lane < 16) ? 0 : 8;

    const unsigned short* A  = hbuf + (size_t)be * CC * HH;
    const unsigned short* Bw = w2t  + (size_t)e * DD * HH;
    const unsigned short* arow0 = A + (size_t)(m0 + lm) * HH + kbase;
    const unsigned short* arow1 = arow0 + (size_t)16 * HH;
    const unsigned short* brow[4];
#pragma unroll
    for (int ni = 0; ni < 4; ++ni)
        brow[ni] = Bw + (size_t)(d0 + ni * 16 + lm) * HH + kbase;

    v8f acc[2][4] = {};
    for (int kk = 0; kk < HH; kk += 32) {
        v16bf a0 = load_frag(arow0 + kk);
        v16bf a1 = load_frag(arow1 + kk);
        v16bf bf[4];
#pragma unroll
        for (int ni = 0; ni < 4; ++ni) bf[ni] = load_frag(brow[ni] + kk);
#pragma unroll
        for (int ni = 0; ni < 4; ++ni) {
            acc[0][ni] = __builtin_amdgcn_wmma_f32_16x16x32_bf16(
                false, a0, false, bf[ni], (short)0, acc[0][ni], false, false);
            acc[1][ni] = __builtin_amdgcn_wmma_f32_16x16x32_bf16(
                false, a1, false, bf[ni], (short)0, acc[1][ni], false, false);
        }
    }
    int rowoff = (lane < 16) ? 0 : 8;
#pragma unroll
    for (int mi = 0; mi < 2; ++mi)
#pragma unroll
        for (int ni = 0; ni < 4; ++ni) {
            int dcol = d0 + ni * 16 + lm;
            float bias = b2[e * DD + dcol];
            int c0 = m0 + mi * 16 + rowoff;
#pragma unroll
            for (int r = 0; r < 8; ++r) {
                int c = c0 + r;
                int sidx = be * CC + c;
                int tok = tok_slot[sidx];
                if (tok >= 0) {
                    float g = slot_gate[sidx];
                    out[((size_t)b * NN + tok) * DD + dcol] = (acc[mi][ni][r] + bias) * g;
                }
            }
        }
}

// ---------------- scalar losses ----------------
__global__ void finalize_k(const int* __restrict__ count, const float* __restrict__ proxy,
                           const float* __restrict__ zz, float* __restrict__ losses) {
    if (threadIdx.x == 0 && blockIdx.x == 0) {
        float aux = 0.0f;
        for (int i = 0; i < BB * EE; ++i)
            aux += ((float)count[i] / (float)NN) * (proxy[i] / (float)NN);
        aux *= (float)(EE * EE) / (float)BB;
        losses[0] = aux;
        losses[1] = (*zz) / (float)(BB * NN);
    }
}

// ---------------- launch ----------------
extern "C" void kernel_launch(void* const* d_in, const int* in_sizes, int n_in,
                              void* d_out, int out_size, void* d_ws, size_t ws_size,
                              hipStream_t stream) {
    (void)in_sizes; (void)n_in; (void)ws_size;
    const float* x    = (const float*)d_in[0];
    const float* wr   = (const float*)d_in[1];
    const float* w1   = (const float*)d_in[2];
    const float* b1   = (const float*)d_in[3];
    const float* w2   = (const float*)d_in[4];
    const float* b2   = (const float*)d_in[5];
    float* out = (float*)d_out;

    // workspace carve-up (256B aligned)
    char* ws = (char*)d_ws;
    size_t off = 0;
    auto take = [&](size_t bytes) -> char* {
        char* p = ws + off;
        off = (off + bytes + 255) & ~(size_t)255;
        return p;
    };
    unsigned short* w1t  = (unsigned short*)take((size_t)EE * HH * DD * 2);   // [E][H][D] bf16
    unsigned short* w2t  = (unsigned short*)take((size_t)EE * DD * HH * 2);   // [E][D][H] bf16
    unsigned short* xe   = (unsigned short*)take((size_t)BB * EE * CC * DD * 2);
    unsigned short* hbuf = (unsigned short*)take((size_t)BB * EE * CC * HH * 2);
    int*   tok_slot  = (int*)  take((size_t)BB * EE * CC * 4);
    float* slot_gate = (float*)take((size_t)BB * EE * CC * 4);
    int*   idx_ws    = (int*)  take((size_t)BB * NN * 4);
    float* gate_ws   = (float*)take((size_t)BB * NN * 4);
    int*   count     = (int*)  take((size_t)BB * EE * 4);
    float* proxy     = (float*)take((size_t)BB * EE * 4);
    float* zz        = (float*)take(4);

    // 1) zero output (dropped tokens must be 0) + init scratch
    zero_f32<<<(out_size + 255) / 256, 256, 0, stream>>>(out, out_size);
    init_ws_k<<<(BB * EE * CC + 255) / 256, 256, 0, stream>>>(tok_slot, proxy, count, zz);

    // 2) weight transpose+convert: w1 [E][D][H] -> w1t [E][H][D]; w2 [E][H][D] -> w2t [E][D][H]
    transpose_cvt<<<EE * (DD / 32) * (HH / 32), 256, 0, stream>>>(w1, w1t, DD, HH);
    transpose_cvt<<<EE * (HH / 32) * (DD / 32), 256, 0, stream>>>(w2, w2t, HH, DD);

    // 3) router: one wave per token (B*N waves, 8 waves/block)
    router_k<<<(BB * NN) / 8, 256, 0, stream>>>(x, wr, idx_ws, gate_ws, proxy, zz);

    // 4) capacity scan: one wave per (b,e)
    scan_k<<<BB * EE, 32, 0, stream>>>(idx_ws, gate_ws, tok_slot, slot_gate, count);

    // 5) gather tokens -> bf16 expert input
    gather_k<<<BB * EE * CC, 256, 0, stream>>>(x, tok_slot, xe);

    // 6) expert FFN layer 1 (WMMA bf16) : waves = 32 * 10 * 64 = 20480 -> 2560 blocks
    gemm1_k<<<(BB * EE * (CC / 32) * (HH / 64)) / 8, 256, 0, stream>>>(xe, w1t, b1, hbuf);

    // 7) expert FFN layer 2 (WMMA bf16) + gated scatter : waves = 32 * 10 * 16 = 5120 -> 640 blocks
    gemm2_k<<<(BB * EE * (CC / 32) * (DD / 64)) / 8, 256, 0, stream>>>(hbuf, w2t, b2, tok_slot,
                                                                       slot_gate, out);

    // 8) losses
    finalize_k<<<1, 32, 0, stream>>>(count, proxy, zz, out + (out_size - 2));
}
